// SwinTransformer_90683939487954
// MI455X (gfx1250) — matrix operand order repaired and checked by reading, not compiled, not run on hardware
//
#include <hip/hip_runtime.h>
#include <hip/hip_bf16.h>
#include <cstdint>
#include <cstddef>

// ---------------------------------------------------------------------------
// Swin block for MI455X (gfx1250): f16 WMMA (v_wmma_f32_16x16x32_f16), wave32.
// Big GEMMs: 16x64 tile per wave (4 accumulators, A-fragment reused 4x).
// ---------------------------------------------------------------------------

typedef __attribute__((ext_vector_type(16))) _Float16 v16h;
typedef __attribute__((ext_vector_type(8)))  float    v8f;

#define WSZ    7
#define NTOK   49
#define NPADT  64
#define HEADS  4
#define HD     32
#define DIMC   128
#define HH     56
#define NWIN   64          // windows per image (8x8)
#define BATCH  64
#define TOKS   (BATCH*HH*HH)   // 200704
#define SHIFTV 3
#define EPSLN  1e-3f

// ---------------------------------------------------------------------------
// Fragment layouts per CDNA5 ISA (CDNA5 ISA §7.12.2):
//   A frag, lane l: row = l&15; elems 0..7 = K(k0+8h..+7), 8..15 = K(k0+16+8h..+7)
//   B frag, lane l: col = l&15 (row of Bt); elems 0..7 = K(k0+8h..+7), 8..15 = K(k0+16+8h..+7)
//   C/D,   lane l: element r -> row r+8h, col l&15      (h = l>>4)
// ---------------------------------------------------------------------------

// Single 16x16 tile (used by attention).
__device__ inline v8f wmma_tile_f16(const _Float16* __restrict__ A, int lda, int row0,
                                    const _Float16* __restrict__ Bt, int ldb, int col0,
                                    int K, int lane) {
  v8f acc = {0.f, 0.f, 0.f, 0.f, 0.f, 0.f, 0.f, 0.f};
  const int mr = lane & 15;
  const int hh = lane >> 4;
  const _Float16* ap = A  + (long)(row0 + mr) * lda + 8  * hh;
  const _Float16* bp = Bt + (long)(col0 + mr) * ldb + 8  * hh;
  for (int k0 = 0; k0 < K; k0 += 32) {
    union { v16h v; float4 f[2]; } ua, ub;
    ua.f[0] = *(const float4*)(ap + k0);
    ua.f[1] = *(const float4*)(ap + k0 + 16);
    ub.f[0] = *(const float4*)(bp + k0);
    ub.f[1] = *(const float4*)(bp + k0 + 16);
    acc = __builtin_amdgcn_wmma_f32_16x16x32_f16(false, ua.v, false, ub.v,
                                                 (short)0, acc, false, false);
  }
  return acc;
}

// 16x64 tile per wave: one A fragment feeds 4 WMMAs (N-tiles col0..col0+63).
__device__ inline void wmma_tile_f16_x4(const _Float16* __restrict__ A, int lda, int row0,
                                        const _Float16* __restrict__ Bt, int ldb, int col0,
                                        int K, int lane, v8f acc[4]) {
  const int mr = lane & 15;
  const int hh = lane >> 4;
  const _Float16* ap = A  + (long)(row0 + mr) * lda + 8  * hh;
  const _Float16* bp = Bt + (long)(col0 + mr) * ldb + 8  * hh;
  for (int k0 = 0; k0 < K; k0 += 32) {
    union { v16h v; float4 f[2]; } ua;
    ua.f[0] = *(const float4*)(ap + k0);
    ua.f[1] = *(const float4*)(ap + k0 + 16);
    // speculative prefetch of the A stream (global_prefetch_b8)
    __builtin_prefetch((const void*)(ap + k0 + 128), 0, 0);
#pragma unroll
    for (int j = 0; j < 4; ++j) {
      union { v16h v; float4 f[2]; } ub;
      const _Float16* bpj = bp + (long)j * 16 * ldb;
      ub.f[0] = *(const float4*)(bpj + k0);
      ub.f[1] = *(const float4*)(bpj + k0 + 16);
      acc[j] = __builtin_amdgcn_wmma_f32_16x16x32_f16(false, ua.v, false, ub.v,
                                                      (short)0, acc[j], false, false);
    }
  }
}

__device__ inline float gelu_exact(float x) {
  return 0.5f * x * (1.0f + erff(x * 0.7071067811865476f));
}

// ---------------------------------------------------------------------------
// Prep kernels
// ---------------------------------------------------------------------------
__global__ void prep_weights_kernel(const float* __restrict__ qkv_w,
                                    const float* __restrict__ proj_w,
                                    const float* __restrict__ fc1_w,
                                    const float* __restrict__ fc2_w,
                                    _Float16* __restrict__ qkvT,
                                    _Float16* __restrict__ projT,
                                    _Float16* __restrict__ fc1T,
                                    _Float16* __restrict__ fc2T) {
  int idx = blockIdx.x * 256 + threadIdx.x;        // up to 65536
  if (idx < 384 * 128) { int n = idx / 128, k = idx % 128; qkvT[idx]  = (_Float16)qkv_w[k * 384 + n]; }
  if (idx < 128 * 128) { int n = idx / 128, k = idx % 128; projT[idx] = (_Float16)proj_w[k * 128 + n]; }
  if (idx < 512 * 128) { int n = idx / 128, k = idx % 128; fc1T[idx]  = (_Float16)fc1_w[k * 512 + n]; }
  if (idx < 128 * 512) { int n = idx / 512, k = idx % 512; fc2T[idx]  = (_Float16)fc2_w[k * 128 + n]; }
}

__global__ void prep_bias_kernel(const float* __restrict__ table, float* __restrict__ biasf) {
  int idx = blockIdx.x * 256 + threadIdx.x;
  if (idx >= HEADS * NTOK * NTOK) return;
  int h = idx / (NTOK * NTOK);
  int nm = idx % (NTOK * NTOK);
  int n = nm / NTOK, m = nm % NTOK;
  // meshgrid 'xy': cf[0][p] = p%7 (x), cf[1][p] = p/7 (y)
  int dx = (n % WSZ) - (m % WSZ) + (WSZ - 1);
  int dy = (n / WSZ) - (m / WSZ) + (WSZ - 1);
  int t = dx * (2 * WSZ - 1) + dy;
  biasf[idx] = table[t * HEADS + h];
}

__device__ inline int swin_region(int v) { return v < HH - WSZ ? 0 : (v < HH - SHIFTV ? 1 : 2); }

__global__ void prep_mask_kernel(float* __restrict__ maskf) {
  int idx = blockIdx.x * 256 + threadIdx.x;
  if (idx >= NWIN * NTOK * NTOK) return;
  int widx = idx / (NTOK * NTOK);
  int nm = idx % (NTOK * NTOK);
  int n = nm / NTOK, m = nm % NTOK;
  int wh = widx >> 3, ww = widx & 7;
  int hn = wh * WSZ + n / WSZ, wn = ww * WSZ + n % WSZ;
  int hm = wh * WSZ + m / WSZ, wm = ww * WSZ + m % WSZ;
  int cn = swin_region(hn) * 3 + swin_region(wn);
  int cm = swin_region(hm) * 3 + swin_region(wm);
  maskf[idx] = (cn == cm) ? 0.0f : -100.0f;
}

__global__ void fill_zero_kernel(float4* __restrict__ p, long n4) {
  long i = (long)blockIdx.x * blockDim.x + threadIdx.x;
  if (i < n4) p[i] = float4{0.f, 0.f, 0.f, 0.f};
}

// ---------------------------------------------------------------------------
// LN1 fused with cyclic shift + window partition.  One wave per window row.
// ---------------------------------------------------------------------------
__global__ __launch_bounds__(256) void ln1_window_kernel(const float* __restrict__ x,
                                                         const float* __restrict__ g,
                                                         const float* __restrict__ b,
                                                         _Float16* __restrict__ xw) {
  int lane = threadIdx.x & 31;
  int wave = threadIdx.x >> 5;
  int r = blockIdx.x * 8 + wave;
  if (r >= TOKS) return;
  int w = r / NTOK, t = r % NTOK;
  int bb = w >> 6, widx = w & 63;
  int wh = widx >> 3, ww = widx & 7;
  int iy = wh * WSZ + t / WSZ + SHIFTV; if (iy >= HH) iy -= HH;
  int ix = ww * WSZ + t % WSZ + SHIFTV; if (ix >= HH) ix -= HH;
  const float* src = x + ((long)bb * (HH * HH) + iy * HH + ix) * DIMC;
  float4 v = ((const float4*)src)[lane];
  float s  = v.x + v.y + v.z + v.w;
  float sq = v.x * v.x + v.y * v.y + v.z * v.z + v.w * v.w;
  for (int o = 16; o > 0; o >>= 1) { s += __shfl_xor(s, o, 32); sq += __shfl_xor(sq, o, 32); }
  float mean = s * (1.f / DIMC);
  float var  = sq * (1.f / DIMC) - mean * mean;
  float inv  = rsqrtf(var + EPSLN);
  float4 gg = ((const float4*)g)[lane];
  float4 bv = ((const float4*)b)[lane];
  _Float16* dst = xw + (long)r * DIMC + lane * 4;
  dst[0] = (_Float16)((v.x - mean) * inv * gg.x + bv.x);
  dst[1] = (_Float16)((v.y - mean) * inv * gg.y + bv.y);
  dst[2] = (_Float16)((v.z - mean) * inv * gg.z + bv.z);
  dst[3] = (_Float16)((v.w - mean) * inv * gg.w + bv.w);
}

// LN2: plain row layernorm, f32 -> f16
__global__ __launch_bounds__(256) void ln2_kernel(const float* __restrict__ x2,
                                                  const float* __restrict__ g,
                                                  const float* __restrict__ b,
                                                  _Float16* __restrict__ xn) {
  int lane = threadIdx.x & 31;
  int wave = threadIdx.x >> 5;
  int r = blockIdx.x * 8 + wave;
  if (r >= TOKS) return;
  const float* src = x2 + (long)r * DIMC;
  float4 v = ((const float4*)src)[lane];
  float s  = v.x + v.y + v.z + v.w;
  float sq = v.x * v.x + v.y * v.y + v.z * v.z + v.w * v.w;
  for (int o = 16; o > 0; o >>= 1) { s += __shfl_xor(s, o, 32); sq += __shfl_xor(sq, o, 32); }
  float mean = s * (1.f / DIMC);
  float var  = sq * (1.f / DIMC) - mean * mean;
  float inv  = rsqrtf(var + EPSLN);
  float4 gg = ((const float4*)g)[lane];
  float4 bv = ((const float4*)b)[lane];
  _Float16* dst = xn + (long)r * DIMC + lane * 4;
  dst[0] = (_Float16)((v.x - mean) * inv * gg.x + bv.x);
  dst[1] = (_Float16)((v.y - mean) * inv * gg.y + bv.y);
  dst[2] = (_Float16)((v.z - mean) * inv * gg.z + bv.z);
  dst[3] = (_Float16)((v.w - mean) * inv * gg.w + bv.w);
}

// ---------------------------------------------------------------------------
// QKV GEMM: [TOKS,128] x [128,384]; 16x64 per wave.  N-group g covers cols
// [g*64, g*64+64) so q/k/v selection (`which`) and `head` are uniform per
// (group, j) -> scalar control flow, no EXEC divergence in the epilogue.
// ---------------------------------------------------------------------------
__global__ __launch_bounds__(256) void gemm_qkv_kernel(const _Float16* __restrict__ A,
                                                       const _Float16* __restrict__ Bt,
                                                       const float* __restrict__ bias,
                                                       _Float16* __restrict__ q,
                                                       _Float16* __restrict__ k,
                                                       _Float16* __restrict__ vT) {
  const int NG = 384 / 64;   // 6 groups
  int lane = threadIdx.x & 31, wave = threadIdx.x >> 5;
  long tile = (long)blockIdx.x * 8 + wave;
  int ng = (int)(tile % NG);
  int mt = (int)(tile / NG);
  v8f acc[4] = {{0.f,0.f,0.f,0.f,0.f,0.f,0.f,0.f}, {0.f,0.f,0.f,0.f,0.f,0.f,0.f,0.f},
                {0.f,0.f,0.f,0.f,0.f,0.f,0.f,0.f}, {0.f,0.f,0.f,0.f,0.f,0.f,0.f,0.f}};
  wmma_tile_f16_x4(A, DIMC, mt * 16, Bt, DIMC, ng * 64, DIMC, lane, acc);
  int hh = lane >> 4;
  int ln = lane & 15;
  // per-r row mapping (independent of j)
  int rows[8], wws[8], tts[8];
#pragma unroll
  for (int r = 0; r < 8; ++r) {
    rows[r] = mt * 16 + r + 8 * hh;
    wws[r] = rows[r] / NTOK;
    tts[r] = rows[r] % NTOK;
  }
#pragma unroll
  for (int j = 0; j < 4; ++j) {
    const int colbase = ng * 64 + j * 16;          // uniform
    const int which = colbase >> 7;                // 0=q 1=k 2=v (uniform)
    const int head  = (colbase >> 5) & 3;          // uniform
    const int hd    = (colbase & 31) + ln;         // 0/16 + lane
    float bv = bias[colbase + ln];
#pragma unroll
    for (int r = 0; r < 8; ++r) {
      long base = ((long)wws[r] * HEADS + head) * (NPADT * HD);
      float val = acc[j][r] + bv;
      if (which == 0)      q[base + tts[r] * HD + hd]     = (_Float16)(val * 0.1767766952966369f);
      else if (which == 1) k[base + tts[r] * HD + hd]     = (_Float16)val;
      else                 vT[base + hd * NPADT + tts[r]] = (_Float16)val;
    }
  }
}

// ---------------------------------------------------------------------------
// Attention: one block (512 thr = 16 waves) per (window, head).
// ---------------------------------------------------------------------------
__global__ __launch_bounds__(512) void attn_kernel(const _Float16* __restrict__ q,
                                                   const _Float16* __restrict__ k,
                                                   const _Float16* __restrict__ vT,
                                                   const float* __restrict__ biasf,
                                                   const float* __restrict__ maskf,
                                                   _Float16* __restrict__ outw) {
  __shared__ __align__(16) float    S[NPADT][NPADT];
  __shared__ __align__(16) _Float16 Pm[NPADT][NPADT];

  int whid = blockIdx.x;             // window*HEADS + head
  long w   = whid >> 2;
  int head = whid & 3;
  int widx = (int)(w & 63);
  int lane = threadIdx.x & 31, wave = threadIdx.x >> 5;

  const _Float16* qb = q  + (long)whid * (NPADT * HD);
  const _Float16* kb = k  + (long)whid * (NPADT * HD);
  const _Float16* vb = vT + (long)whid * (NPADT * HD);

  // Phase 1: S = q k^T (4x4 tiles of 16x16, K=32 -> single WMMA per tile)
  {
    int ti = wave >> 2, tj = wave & 3;
    v8f acc = wmma_tile_f16(qb, HD, ti * 16, kb, HD, tj * 16, HD, lane);
    int n = tj * 16 + (lane & 15);
    int hh = lane >> 4;
    const float* bptr = biasf + head * (NTOK * NTOK);
    const float* mptr = maskf + widx * (NTOK * NTOK);
#pragma unroll
    for (int r = 0; r < 8; ++r) {
      int m = ti * 16 + r + 8 * hh;
      float val = -1e30f;
      if (m < NTOK && n < NTOK)
        val = acc[r] + bptr[m * NTOK + n] + mptr[m * NTOK + n];
      S[m][n] = val;
    }
  }
  __syncthreads();

  // Phase 2: softmax applied twice (second pass over the 49 real columns only)
  {
    int row = threadIdx.x >> 3;     // 0..63
    int sub = threadIdx.x & 7;      // 8 threads per row
    float v[8], e[8];
#pragma unroll
    for (int kk = 0; kk < 8; ++kk) v[kk] = S[row][kk * 8 + sub];
    float mx = -3e30f;
#pragma unroll
    for (int kk = 0; kk < 8; ++kk) mx = fmaxf(mx, v[kk]);
    for (int o = 4; o > 0; o >>= 1) mx = fmaxf(mx, __shfl_xor(mx, o, 8));
    float sum = 0.f;
#pragma unroll
    for (int kk = 0; kk < 8; ++kk) { e[kk] = expf(v[kk] - mx); sum += e[kk]; }
    for (int o = 4; o > 0; o >>= 1) sum += __shfl_xor(sum, o, 8);
    float inv = 1.f / sum;
#pragma unroll
    for (int kk = 0; kk < 8; ++kk) v[kk] = e[kk] * inv;      // p1 (pads -> 0)
    // second softmax, valid columns only
    float mx2 = -3e30f;
#pragma unroll
    for (int kk = 0; kk < 8; ++kk) { int c = kk * 8 + sub; if (c < NTOK) mx2 = fmaxf(mx2, v[kk]); }
    for (int o = 4; o > 0; o >>= 1) mx2 = fmaxf(mx2, __shfl_xor(mx2, o, 8));
    float sum2 = 0.f;
#pragma unroll
    for (int kk = 0; kk < 8; ++kk) {
      int c = kk * 8 + sub;
      e[kk] = (c < NTOK) ? expf(v[kk] - mx2) : 0.f;
      sum2 += e[kk];
    }
    for (int o = 4; o > 0; o >>= 1) sum2 += __shfl_xor(sum2, o, 8);
    float inv2 = 1.f / sum2;
#pragma unroll
    for (int kk = 0; kk < 8; ++kk) Pm[row][kk * 8 + sub] = (_Float16)(e[kk] * inv2);
  }
  __syncthreads();

  // Phase 3: out = P @ V  (M=64 pad, N=32, K=64 -> 2 WMMAs per tile; 8 waves)
  if (wave < 8) {
    int ti = wave >> 1, tj = wave & 1;
    v8f acc = wmma_tile_f16(&Pm[0][0], NPADT, ti * 16, vb, NPADT, tj * 16, NPADT, lane);
    int n = tj * 16 + (lane & 15);
    int hh = lane >> 4;
#pragma unroll
    for (int r = 0; r < 8; ++r) {
      int m = ti * 16 + r + 8 * hh;
      if (m < NTOK)
        outw[((long)w * NTOK + m) * DIMC + head * HD + n] = (_Float16)acc[r];
    }
  }
}

// ---------------------------------------------------------------------------
// Proj GEMM (16x64/wave) fused with window-reverse + un-shift + residual -> x2
// ---------------------------------------------------------------------------
__global__ __launch_bounds__(256) void gemm_proj_kernel(const _Float16* __restrict__ A,
                                                        const _Float16* __restrict__ Bt,
                                                        const float* __restrict__ bias,
                                                        const float* __restrict__ xin,
                                                        float* __restrict__ x2) {
  const int NG = DIMC / 64;  // 2
  int lane = threadIdx.x & 31, wave = threadIdx.x >> 5;
  long tile = (long)blockIdx.x * 8 + wave;
  int ng = (int)(tile % NG);
  int mt = (int)(tile / NG);
  v8f acc[4] = {{0.f,0.f,0.f,0.f,0.f,0.f,0.f,0.f}, {0.f,0.f,0.f,0.f,0.f,0.f,0.f,0.f},
                {0.f,0.f,0.f,0.f,0.f,0.f,0.f,0.f}, {0.f,0.f,0.f,0.f,0.f,0.f,0.f,0.f}};
  wmma_tile_f16_x4(A, DIMC, mt * 16, Bt, DIMC, ng * 64, DIMC, lane, acc);
  int hh = lane >> 4;
  int ln = lane & 15;
  long dstrow[8];
#pragma unroll
  for (int r = 0; r < 8; ++r) {
    int row = mt * 16 + r + 8 * hh;          // window-ordered row
    int w = row / NTOK, t = row % NTOK;
    int bb = w >> 6, widx = w & 63;
    int wh = widx >> 3, ww = widx & 7;
    int iy = wh * WSZ + t / WSZ + SHIFTV; if (iy >= HH) iy -= HH;
    int ix = ww * WSZ + t % WSZ + SHIFTV; if (ix >= HH) ix -= HH;
    dstrow[r] = ((long)bb * (HH * HH) + iy * HH + ix) * DIMC;
  }
#pragma unroll
  for (int j = 0; j < 4; ++j) {
    int col = ng * 64 + j * 16 + ln;
    float bv = bias[col];
#pragma unroll
    for (int r = 0; r < 8; ++r) {
      long dst = dstrow[r] + col;
      x2[dst] = xin[dst] + acc[j][r] + bv;
    }
  }
}

// ---------------------------------------------------------------------------
// FC1 GEMM (16x64/wave) + bias + gelu(gelu(x)) -> h1 f16 [TOKS,512]
// ---------------------------------------------------------------------------
__global__ __launch_bounds__(256) void gemm_fc1_kernel(const _Float16* __restrict__ A,
                                                       const _Float16* __restrict__ Bt,
                                                       const float* __restrict__ bias,
                                                       _Float16* __restrict__ h1) {
  const int NG = 512 / 64;  // 8
  int lane = threadIdx.x & 31, wave = threadIdx.x >> 5;
  long tile = (long)blockIdx.x * 8 + wave;
  int ng = (int)(tile % NG);
  int mt = (int)(tile / NG);
  v8f acc[4] = {{0.f,0.f,0.f,0.f,0.f,0.f,0.f,0.f}, {0.f,0.f,0.f,0.f,0.f,0.f,0.f,0.f},
                {0.f,0.f,0.f,0.f,0.f,0.f,0.f,0.f}, {0.f,0.f,0.f,0.f,0.f,0.f,0.f,0.f}};
  wmma_tile_f16_x4(A, DIMC, mt * 16, Bt, DIMC, ng * 64, DIMC, lane, acc);
  int hh = lane >> 4;
  int ln = lane & 15;
#pragma unroll
  for (int j = 0; j < 4; ++j) {
    int col = ng * 64 + j * 16 + ln;
    float bv = bias[col];
#pragma unroll
    for (int r = 0; r < 8; ++r) {
      int row = mt * 16 + r + 8 * hh;
      float val = acc[j][r] + bv;
      val = gelu_exact(gelu_exact(val));
      h1[(long)row * 512 + col] = (_Float16)val;
    }
  }
}

// ---------------------------------------------------------------------------
// FC2 GEMM (16x64/wave, K=512) + bias + residual -> d_out (f32)
// ---------------------------------------------------------------------------
__global__ __launch_bounds__(256) void gemm_fc2_kernel(const _Float16* __restrict__ A,
                                                       const _Float16* __restrict__ Bt,
                                                       const float* __restrict__ bias,
                                                       const float* __restrict__ x2,
                                                       float* __restrict__ outp) {
  const int NG = DIMC / 64;  // 2
  int lane = threadIdx.x & 31, wave = threadIdx.x >> 5;
  long tile = (long)blockIdx.x * 8 + wave;
  int ng = (int)(tile % NG);
  int mt = (int)(tile / NG);
  v8f acc[4] = {{0.f,0.f,0.f,0.f,0.f,0.f,0.f,0.f}, {0.f,0.f,0.f,0.f,0.f,0.f,0.f,0.f},
                {0.f,0.f,0.f,0.f,0.f,0.f,0.f,0.f}, {0.f,0.f,0.f,0.f,0.f,0.f,0.f,0.f}};
  wmma_tile_f16_x4(A, 512, mt * 16, Bt, 512, ng * 64, 512, lane, acc);
  int hh = lane >> 4;
  int ln = lane & 15;
#pragma unroll
  for (int j = 0; j < 4; ++j) {
    int col = ng * 64 + j * 16 + ln;
    float bv = bias[col];
#pragma unroll
    for (int r = 0; r < 8; ++r) {
      int row = mt * 16 + r + 8 * hh;
      long idx = (long)row * DIMC + col;
      outp[idx] = x2[idx] + acc[j][r] + bv;
    }
  }
}

// ---------------------------------------------------------------------------
// Host launch
// ---------------------------------------------------------------------------
extern "C" void kernel_launch(void* const* d_in, const int* in_sizes, int n_in,
                              void* d_out, int out_size, void* d_ws, size_t ws_size,
                              hipStream_t stream) {
  (void)in_sizes; (void)n_in; (void)out_size; (void)ws_size;
  const float* x      = (const float*)d_in[0];
  const float* n1g    = (const float*)d_in[1];
  const float* n1b    = (const float*)d_in[2];
  const float* qkv_w  = (const float*)d_in[3];
  const float* qkv_b  = (const float*)d_in[4];
  const float* btab   = (const float*)d_in[5];
  const float* proj_w = (const float*)d_in[6];
  const float* proj_b = (const float*)d_in[7];
  const float* n2g    = (const float*)d_in[8];
  const float* n2b    = (const float*)d_in[9];
  const float* fc1_w  = (const float*)d_in[10];
  const float* fc1_b  = (const float*)d_in[11];
  const float* fc2_w  = (const float*)d_in[12];
  const float* fc2_b  = (const float*)d_in[13];
  float* outp = (float*)d_out;
  char* ws = (char*)d_ws;

  // Workspace layout (bytes, 256-aligned)
  const size_t SZ_XW  = (size_t)TOKS * DIMC * 2;                 // 51,380,224
  const size_t SZ_QKV1 = (size_t)4096 * HEADS * NPADT * HD * 2;  // 67,108,864 each
  const size_t SZ_H1  = (size_t)TOKS * 512 * 2;                  // 205,520,896
  const size_t OFF_XW   = 0;                       // xw f16; reused as attn_out
  const size_t OFF_QKV  = OFF_XW + SZ_XW;          // q,k,vT; reused as h1
  const size_t OFF_Q    = OFF_QKV;
  const size_t OFF_K    = OFF_QKV + SZ_QKV1;
  const size_t OFF_V    = OFF_QKV + 2 * SZ_QKV1;
  const size_t OFF_X2   = OFF_QKV + SZ_H1;         // f32 [TOKS,128]
  const size_t OFF_X2N  = OFF_X2 + (size_t)TOKS * DIMC * 4;
  const size_t OFF_WQ   = OFF_X2N + SZ_XW;
  const size_t OFF_WP   = OFF_WQ + 384 * 128 * 2;
  const size_t OFF_W1   = OFF_WP + 128 * 128 * 2;
  const size_t OFF_W2   = OFF_W1 + 512 * 128 * 2;
  const size_t OFF_BIAS = OFF_W2 + 128 * 512 * 2;
  const size_t OFF_MASK = OFF_BIAS + 38656;

  _Float16* xw    = (_Float16*)(ws + OFF_XW);
  _Float16* attn_o= (_Float16*)(ws + OFF_XW);      // alias (xw dead after QKV GEMM)
  _Float16* qbuf  = (_Float16*)(ws + OFF_Q);
  _Float16* kbuf  = (_Float16*)(ws + OFF_K);
  _Float16* vbuf  = (_Float16*)(ws + OFF_V);
  _Float16* h1    = (_Float16*)(ws + OFF_QKV);     // alias (q/k/v dead after attn)
  float*    x2    = (float*)(ws + OFF_X2);
  _Float16* x2n   = (_Float16*)(ws + OFF_X2N);
  _Float16* qkvT  = (_Float16*)(ws + OFF_WQ);
  _Float16* projT = (_Float16*)(ws + OFF_WP);
  _Float16* fc1T  = (_Float16*)(ws + OFF_W1);
  _Float16* fc2T  = (_Float16*)(ws + OFF_W2);
  float*    biasf = (float*)(ws + OFF_BIAS);
  float*    maskf = (float*)(ws + OFF_MASK);

  // 0) prep: weight transposes (f16), rel-pos bias gather, shift mask, zero pads
  hipLaunchKernelGGL(prep_weights_kernel, dim3(256), dim3(256), 0, stream,
                     qkv_w, proj_w, fc1_w, fc2_w, qkvT, projT, fc1T, fc2T);
  hipLaunchKernelGGL(prep_bias_kernel, dim3((HEADS * NTOK * NTOK + 255) / 256), dim3(256),
                     0, stream, btab, biasf);
  hipLaunchKernelGGL(prep_mask_kernel, dim3((NWIN * NTOK * NTOK + 255) / 256), dim3(256),
                     0, stream, maskf);
  {
    long n4 = (long)(3 * SZ_QKV1) / 16;
    hipLaunchKernelGGL(fill_zero_kernel, dim3((unsigned)((n4 + 255) / 256)), dim3(256),
                       0, stream, (float4*)(ws + OFF_QKV), n4);
  }

  // 1) LN1 + shift + window partition
  hipLaunchKernelGGL(ln1_window_kernel, dim3(TOKS / 8), dim3(256), 0, stream,
                     x, n1g, n1b, xw);

  // 2) QKV GEMM (M=200704, N=384, K=128) -> q (scaled), k, v^T
  hipLaunchKernelGGL(gemm_qkv_kernel, dim3((TOKS / 16) * (384 / 64) / 8), dim3(256),
                     0, stream, xw, qkvT, qkv_b, qbuf, kbuf, vbuf);

  // 3) windowed attention (16384 blocks x 512 threads)
  hipLaunchKernelGGL(attn_kernel, dim3(4096 * HEADS), dim3(512), 0, stream,
                     qbuf, kbuf, vbuf, biasf, maskf, attn_o);

  // 4) proj GEMM + window reverse + un-shift + residual -> x2
  hipLaunchKernelGGL(gemm_proj_kernel, dim3((TOKS / 16) * (DIMC / 64) / 8), dim3(256),
                     0, stream, attn_o, projT, proj_b, x, x2);

  // 5) LN2 -> f16
  hipLaunchKernelGGL(ln2_kernel, dim3(TOKS / 8), dim3(256), 0, stream, x2, n2g, n2b, x2n);

  // 6) FC1 GEMM + gelu^2 -> h1
  hipLaunchKernelGGL(gemm_fc1_kernel, dim3((TOKS / 16) * (512 / 64) / 8), dim3(256),
                     0, stream, x2n, fc1T, fc1_b, h1);

  // 7) FC2 GEMM + residual -> d_out
  hipLaunchKernelGGL(gemm_fc2_kernel, dim3((TOKS / 16) * (DIMC / 64) / 8), dim3(256),
                     0, stream, h1, fc2T, fc2_b, x2, outp);
}
